// MultiHeadedCausalSelfAttention_16466904613086
// MI455X (gfx1250) — matrix operand order repaired and checked by reading, not compiled, not run on hardware
//
#include <hip/hip_runtime.h>

typedef __attribute__((ext_vector_type(16))) __bf16 v16bf;
typedef __attribute__((ext_vector_type(8)))  __bf16 v8bf;
typedef __attribute__((ext_vector_type(8)))  float  v8f;

union V16U { v16bf v; v8bf h[2]; };

static __device__ __forceinline__ __bf16 f2bf(float f) {
  unsigned u = __builtin_bit_cast(unsigned, f);
  u += 0x7FFFu + ((u >> 16) & 1u);          // round-to-nearest-even
  unsigned short s = (unsigned short)(u >> 16);
  return __builtin_bit_cast(__bf16, s);
}
static __device__ __forceinline__ float bf2f(__bf16 b) {
  unsigned short s = __builtin_bit_cast(unsigned short, b);
  return __builtin_bit_cast(float, (unsigned)s << 16);
}

// ---------------- f32 -> bf16 conversion ----------------
__global__ void __launch_bounds__(256) cvt_f32_bf16(const float* __restrict__ in,
                                                    __bf16* __restrict__ out, int n) {
  int i = blockIdx.x * 256 + threadIdx.x;
  if (i < n) out[i] = f2bf(in[i]);
}

// ---------------- GEMM: C[M,N] = A[M,K] @ W[N,K]^T (bf16 WMMA, f32 acc) ----------------
// One wave computes a 32(M) x 64(N) tile: 8 accumulators, 2 A-operands x 4 B-operands
// per K-step of 32 -> 8 WMMAs per 12 b128 loads.  Block = 8 waves arranged
// 4(M) x 2(N) -> 128 x 128 block tile.
__global__ void __launch_bounds__(256) gemm_bf16_wmma(
    const __bf16* __restrict__ A, const __bf16* __restrict__ W,
    __bf16* __restrict__ Cb, float* __restrict__ Cf, int M, int N, int K) {
  const int lane = threadIdx.x & 31;
  const int wid  = threadIdx.x >> 5;
  const int half = lane >> 4;
  const int l15  = lane & 15;
  const int m0 = blockIdx.y * 128 + (wid >> 1) * 32;
  const int n0 = blockIdx.x * 128 + (wid & 1) * 64;

  v8f acc[2][4] = {};
  const __bf16* ar0 = A + (size_t)(m0 + l15) * K;
  const __bf16* ar1 = A + (size_t)(m0 + 16 + l15) * K;
  const __bf16* wr0 = W + (size_t)(n0 +  0 + l15) * K;
  const __bf16* wr1 = W + (size_t)(n0 + 16 + l15) * K;
  const __bf16* wr2 = W + (size_t)(n0 + 32 + l15) * K;
  const __bf16* wr3 = W + (size_t)(n0 + 48 + l15) * K;

  for (int kk = 0; kk < K; kk += 32) {
    V16U a0, a1;
    a0.h[0] = *(const v8bf*)(ar0 + kk + half * 16);       // K = kk + 16*half + 0..7
    a0.h[1] = *(const v8bf*)(ar0 + kk + 8 + half * 16);   // K = kk + 16*half + 8..15
    a1.h[0] = *(const v8bf*)(ar1 + kk + half * 16);
    a1.h[1] = *(const v8bf*)(ar1 + kk + 8 + half * 16);
    const int kb = kk + half * 16;                        // B: K = kk + 16*half + 0..15
    v16bf b0 = *(const v16bf*)(wr0 + kb);
    v16bf b1 = *(const v16bf*)(wr1 + kb);
    v16bf b2 = *(const v16bf*)(wr2 + kb);
    v16bf b3 = *(const v16bf*)(wr3 + kb);
    acc[0][0] = __builtin_amdgcn_wmma_f32_16x16x32_bf16(false, a0.v, false, b0, (short)0, acc[0][0], false, false);
    acc[1][0] = __builtin_amdgcn_wmma_f32_16x16x32_bf16(false, a1.v, false, b0, (short)0, acc[1][0], false, false);
    acc[0][1] = __builtin_amdgcn_wmma_f32_16x16x32_bf16(false, a0.v, false, b1, (short)0, acc[0][1], false, false);
    acc[1][1] = __builtin_amdgcn_wmma_f32_16x16x32_bf16(false, a1.v, false, b1, (short)0, acc[1][1], false, false);
    acc[0][2] = __builtin_amdgcn_wmma_f32_16x16x32_bf16(false, a0.v, false, b2, (short)0, acc[0][2], false, false);
    acc[1][2] = __builtin_amdgcn_wmma_f32_16x16x32_bf16(false, a1.v, false, b2, (short)0, acc[1][2], false, false);
    acc[0][3] = __builtin_amdgcn_wmma_f32_16x16x32_bf16(false, a0.v, false, b3, (short)0, acc[0][3], false, false);
    acc[1][3] = __builtin_amdgcn_wmma_f32_16x16x32_bf16(false, a1.v, false, b3, (short)0, acc[1][3], false, false);
  }

  #pragma unroll
  for (int mt = 0; mt < 2; ++mt) {
    #pragma unroll
    for (int j = 0; j < 4; ++j) {
      #pragma unroll
      for (int r = 0; r < 8; ++r) {
        int mr = m0 + mt * 16 + r + 8 * half;  // C/D layout: VGPR r -> row r + 8*half
        int nc = n0 + j * 16 + l15;
        float v = acc[mt][j][r];
        if (Cf) Cf[(size_t)mr * N + nc] = v;
        else    Cb[(size_t)mr * N + nc] = f2bf(v);
      }
    }
  }
}

// ---------------- RoPE in place on bf16 [B*T, 1024] (head-contiguous) ----------------
__global__ void __launch_bounds__(256) rope_inplace(__bf16* __restrict__ qk,
                                                    const int* __restrict__ pos, int total) {
  int i = blockIdx.x * 256 + threadIdx.x;
  if (i >= total) return;                 // total = B*T*H*32
  int j = i & 31;                         // pair index within head (d/2 = 32)
  int tmp = i >> 5;
  int h = tmp & 15;  tmp >>= 4;
  int t = tmp & 2047;
  int b = tmp >> 11;
  float p = (float)pos[t];
  float inv = __powf(10000.0f, -(float)(2 * j) * (1.0f / 64.0f));
  float s, c;
  __sincosf(p * inv, &s, &c);
  size_t base = ((size_t)(b * 2048 + t)) * 1024 + h * 64 + 2 * j;
  float x1 = bf2f(qk[base]);
  float x2 = bf2f(qk[base + 1]);
  qk[base]     = f2bf(c * x1 - s * x2);
  qk[base + 1] = f2bf(s * x1 + c * x2);
}

// ---------------- V transpose: [B*T, H*64] -> [B,H,64,T] ----------------
__global__ void __launch_bounds__(256) v_transpose(const __bf16* __restrict__ v,
                                                   __bf16* __restrict__ vt, int total) {
  int i = blockIdx.x * 256 + threadIdx.x;
  if (i >= total) return;                 // total = B*T*1024
  int e = i & 1023;
  int tmp = i >> 10;
  int t = tmp & 2047;
  int b = tmp >> 11;
  int h = e >> 6;
  int d = e & 63;
  vt[(((size_t)(b * 16 + h) * 64 + d) << 11) + t] = v[i];
}

// ---------------- Flash attention: one wave per 16-row q-tile ----------------
struct __align__(32) AttnTile {
  float  S[16][32];
  __bf16 P[16][32];
  float  scl[16];
};

__global__ void __launch_bounds__(256) flash_attn(
    const __bf16* __restrict__ Q,   // [B*T, 1024] rope'd
    const __bf16* __restrict__ Km,  // [B*T, 1024] rope'd
    const __bf16* __restrict__ Vt,  // [B,H,64,T]
    __bf16* __restrict__ O) {       // [B*T, 1024]
  __shared__ AttnTile tiles[8];
  const int lane = threadIdx.x & 31;
  const int wv   = threadIdx.x >> 5;
  AttnTile& tl = tiles[wv];
  const int half = lane >> 4, l15 = lane & 15;
  int wid = blockIdx.x * 8 + wv;          // 4096 waves total
  int qi  = wid & 127;                    // T/16 = 128 q-tiles per (b,h)
  int bh  = wid >> 7;
  int h = bh & 15, b = bh >> 4;
  const int qb0 = qi * 16;

  // Q A-operand: 16x64 split into two 16x32 chunks
  const __bf16* qrow = Q + ((size_t)(b * 2048 + qb0 + l15)) * 1024 + h * 64;
  V16U aq0, aq1;
  aq0.h[0] = *(const v8bf*)(qrow + half * 16);
  aq0.h[1] = *(const v8bf*)(qrow + 8 + half * 16);
  aq1.h[0] = *(const v8bf*)(qrow + 32 + half * 16);
  aq1.h[1] = *(const v8bf*)(qrow + 40 + half * 16);

  v8f o0 = {}, o1 = {}, o2 = {}, o3 = {};
  float mrun = -1e30f, lrun = 0.0f;

  const __bf16* kbase = Km + ((size_t)(b * 2048)) * 1024 + h * 64;
  const __bf16* vtb   = Vt + ((size_t)(b * 16 + h)) * 64 * 2048;

  for (int kbs = 0; kbs < qb0 + 16; kbs += 32) {
    // S = (Q K^T) for 32 keys: two 16-wide n-tiles, K-dim 64 = 2 chunks
    v8f s0 = {}, s1 = {};
    {
      const __bf16* kr0 = kbase + (size_t)(kbs + l15) * 1024;
      const __bf16* kr1 = kbase + (size_t)(kbs + 16 + l15) * 1024;
      v16bf b00 = *(const v16bf*)(kr0 + half * 16);
      v16bf b01 = *(const v16bf*)(kr0 + 32 + half * 16);
      v16bf b10 = *(const v16bf*)(kr1 + half * 16);
      v16bf b11 = *(const v16bf*)(kr1 + 32 + half * 16);
      s0 = __builtin_amdgcn_wmma_f32_16x16x32_bf16(false, aq0.v, false, b00, (short)0, s0, false, false);
      s0 = __builtin_amdgcn_wmma_f32_16x16x32_bf16(false, aq1.v, false, b01, (short)0, s0, false, false);
      s1 = __builtin_amdgcn_wmma_f32_16x16x32_bf16(false, aq0.v, false, b10, (short)0, s1, false, false);
      s1 = __builtin_amdgcn_wmma_f32_16x16x32_bf16(false, aq1.v, false, b11, (short)0, s1, false, false);
    }
    // scale + causal mask, spill S tile to this wave's LDS slab
    #pragma unroll
    for (int r = 0; r < 8; ++r) {
      int tq  = qb0 + r + 8 * half;
      float v0 = s0[r] * 0.125f;          // 1/sqrt(64)
      float v1 = s1[r] * 0.125f;
      if (kbs + l15 > tq)      v0 = -1e30f;
      if (kbs + 16 + l15 > tq) v1 = -1e30f;
      tl.S[r + 8 * half][l15]      = v0;
      tl.S[r + 8 * half][16 + l15] = v1;
    }
    __builtin_amdgcn_wave_barrier();
    // online softmax: lanes 0..15 own one row each
    if (lane < 16) {
      const int m = lane;
      float row[32];
      float mx = mrun;
      #pragma unroll
      for (int i2 = 0; i2 < 32; ++i2) { row[i2] = tl.S[m][i2]; mx = fmaxf(mx, row[i2]); }
      float alpha = __expf(mrun - mx);
      float sum = 0.0f;
      #pragma unroll
      for (int i2 = 0; i2 < 32; ++i2) {
        float pv = __expf(row[i2] - mx);
        sum += pv;
        tl.P[m][i2] = f2bf(pv);
      }
      lrun = lrun * alpha + sum;
      mrun = mx;
      tl.scl[m] = alpha;
    }
    __builtin_amdgcn_wave_barrier();
    // rescale O accumulators by per-row alpha
    #pragma unroll
    for (int r = 0; r < 8; ++r) {
      float al = tl.scl[r + 8 * half];
      o0[r] *= al; o1[r] *= al; o2[r] *= al; o3[r] *= al;
    }
    // P as A-operand (16x32), V^T columns as B-operand
    V16U ap;
    ap.h[0] = *(const v8bf*)(&tl.P[l15][half * 16]);
    ap.h[1] = *(const v8bf*)(&tl.P[l15][8 + half * 16]);
    v16bf vb0 = *(const v16bf*)(vtb + (size_t)( 0 + l15) * 2048 + kbs + half * 16);
    v16bf vb1 = *(const v16bf*)(vtb + (size_t)(16 + l15) * 2048 + kbs + half * 16);
    v16bf vb2 = *(const v16bf*)(vtb + (size_t)(32 + l15) * 2048 + kbs + half * 16);
    v16bf vb3 = *(const v16bf*)(vtb + (size_t)(48 + l15) * 2048 + kbs + half * 16);
    o0 = __builtin_amdgcn_wmma_f32_16x16x32_bf16(false, ap.v, false, vb0, (short)0, o0, false, false);
    o1 = __builtin_amdgcn_wmma_f32_16x16x32_bf16(false, ap.v, false, vb1, (short)0, o1, false, false);
    o2 = __builtin_amdgcn_wmma_f32_16x16x32_bf16(false, ap.v, false, vb2, (short)0, o2, false, false);
    o3 = __builtin_amdgcn_wmma_f32_16x16x32_bf16(false, ap.v, false, vb3, (short)0, o3, false, false);
    __builtin_amdgcn_wave_barrier();
  }

  // final 1/l normalization and store (bf16, head-contiguous [B*T, 1024])
  if (lane < 16) tl.scl[lane] = 1.0f / lrun;
  __builtin_amdgcn_wave_barrier();
  #pragma unroll
  for (int r = 0; r < 8; ++r) {
    float inv = tl.scl[r + 8 * half];
    size_t rb = ((size_t)(b * 2048 + qb0 + r + 8 * half)) * 1024 + h * 64;
    O[rb +  0 + l15] = f2bf(o0[r] * inv);
    O[rb + 16 + l15] = f2bf(o1[r] * inv);
    O[rb + 32 + l15] = f2bf(o2[r] * inv);
    O[rb + 48 + l15] = f2bf(o3[r] * inv);
  }
}

// ---------------- host-side orchestration ----------------
extern "C" void kernel_launch(void* const* d_in, const int* in_sizes, int n_in,
                              void* d_out, int out_size, void* d_ws, size_t ws_size,
                              hipStream_t stream) {
  (void)in_sizes; (void)n_in; (void)out_size; (void)ws_size;
  const float* x  = (const float*)d_in[0];
  const int*  pos = (const int*)d_in[1];
  const float* Wq = (const float*)d_in[2];
  const float* Wk = (const float*)d_in[3];
  const float* Wv = (const float*)d_in[4];
  const float* Wo = (const float*)d_in[5];
  float* out = (float*)d_out;

  const int Bv = 2, T = 2048, D = 1024, H = 16;
  const int M = Bv * T;                     // 4096

  char* p = (char*)d_ws;
  auto alloc = [&](size_t bytes) -> char* {
    char* r = p;
    p += (bytes + 255) & ~(size_t)255;
    return r;
  };
  __bf16* xb  = (__bf16*)alloc((size_t)M * D * 2);
  __bf16* wqb = (__bf16*)alloc((size_t)D * D * 2);
  __bf16* wkb = (__bf16*)alloc((size_t)D * D * 2);
  __bf16* wvb = (__bf16*)alloc((size_t)D * D * 2);
  __bf16* wob = (__bf16*)alloc((size_t)D * D * 2);
  __bf16* qb  = (__bf16*)alloc((size_t)M * D * 2);
  __bf16* kb  = (__bf16*)alloc((size_t)M * D * 2);
  __bf16* vb  = (__bf16*)alloc((size_t)M * D * 2);
  __bf16* vt  = (__bf16*)alloc((size_t)Bv * H * 64 * T * 2);
  __bf16* ab  = (__bf16*)alloc((size_t)M * D * 2);

  // 1) convert inputs to bf16
  cvt_f32_bf16<<<(M * D + 255) / 256, 256, 0, stream>>>(x,  xb,  M * D);
  cvt_f32_bf16<<<(D * D + 255) / 256, 256, 0, stream>>>(Wq, wqb, D * D);
  cvt_f32_bf16<<<(D * D + 255) / 256, 256, 0, stream>>>(Wk, wkb, D * D);
  cvt_f32_bf16<<<(D * D + 255) / 256, 256, 0, stream>>>(Wv, wvb, D * D);
  cvt_f32_bf16<<<(D * D + 255) / 256, 256, 0, stream>>>(Wo, wob, D * D);

  // 2) projections q,k,v = x @ W^T   (block tile 128x128, wave tile 32x64)
  dim3 gg(D / 128, M / 128);
  gemm_bf16_wmma<<<gg, 256, 0, stream>>>(xb, wqb, qb, nullptr, M, D, D);
  gemm_bf16_wmma<<<gg, 256, 0, stream>>>(xb, wkb, kb, nullptr, M, D, D);
  gemm_bf16_wmma<<<gg, 256, 0, stream>>>(xb, wvb, vb, nullptr, M, D, D);

  // 3) RoPE on q and k (in place)
  int nro = Bv * T * H * 32;
  rope_inplace<<<(nro + 255) / 256, 256, 0, stream>>>(qb, pos, nro);
  rope_inplace<<<(nro + 255) / 256, 256, 0, stream>>>(kb, pos, nro);

  // 4) V -> V^T [B,H,64,T]
  v_transpose<<<(M * D + 255) / 256, 256, 0, stream>>>(vb, vt, M * D);

  // 5) flash attention -> ab
  flash_attn<<<(Bv * H * (T / 16)) / 8, 256, 0, stream>>>(qb, kb, vt, ab);

  // 6) output projection -> f32 d_out
  gemm_bf16_wmma<<<gg, 256, 0, stream>>>(ab, wob, nullptr, out, M, D, D);
}
